// PatchDownsampleBlock_25262997635394
// MI455X (gfx1250) — compile-verified, matrix operand order checked by reading, not run on hardware
//
#include <hip/hip_runtime.h>
#include <math.h>

// ---------------------------------------------------------------------------
// PatchDownsampleBlock fused for MI455X (gfx1250, wave32, WMMA bf16 16x16x32)
//
// Math restructure (exact up to bf16 rounding of operands):
//   out_w = (sum_i a_i x_i) @ Wv + bv            (softmax weights sum to 1)
//   q.k_i = m P x_i^T + x_i.u + m.u2 + t3,  P = Wq Wk^T, u = Wk bq, u2 = Wq bk
// => 4 GEMMs, all M=32768: means@P, xbar@Wv, out@W1(+GELU), hdn@W2(+LN).
// ~58 GFLOP vs ~250 MB HBM traffic -> memory-bound (~11us floor @ 23.3 TB/s),
// so bf16 WMMA with f32 accumulation keeps compute comfortably hidden.
// ---------------------------------------------------------------------------

typedef __attribute__((ext_vector_type(16))) __bf16        v16bf;
typedef __attribute__((ext_vector_type(8)))  float         v8f;
typedef __attribute__((ext_vector_type(4)))  unsigned int  v4u;
typedef __attribute__((ext_vector_type(8)))  unsigned int  v8u;
typedef __attribute__((ext_vector_type(4)))  float         v4f;
typedef unsigned short u16;

#define D_IN   384
#define D_OUT  384
#define D_HID  768
#define NWIN   32768        // (B*T) * (16*16)
#define WPB    16           // windows per workgroup
#define NWG    (NWIN / WPB) // 2048
#define SCALE_QK 0.051031036307982884f   // 384^-0.5

// LDS element strides (padded)
#define KS384  392          // bf16 row stride for 384-wide rows
#define KS768  776          // bf16 row stride for 768-wide rows
#define FS384  388          // f32 row stride for 384-wide rows

// LDS byte offsets
#define OFF_XTOK  0          // 64 x KS384 bf16 = 50176 B (tokens)
#define OFF_XBAR  50176      // 16 x KS384 bf16 = 12544 B (means, then xbar)
#define OFF_POUT  62720      // 16 x FS384 f32  = 24832 B (p = m@P + u)
#define OFF_OUTR  87552      // 16 x KS384 bf16 = 12544 B (attn out)
#define OFF_CW    100096     // 16 f32
#define LDS_BYTES 100352
// overlays (XTOK dead after phase 4):
#define OFF_HDN   0          // 16 x KS768 bf16 = 24832 B
#define OFF_YROW  24832      // 16 x FS384 f32  = 24832 B

static __device__ __forceinline__ u16 f2bf(float f) {
  unsigned int u = __float_as_uint(f);
  u += 0x7FFFu + ((u >> 16) & 1u);          // round-to-nearest-even
  return (u16)(u >> 16);
}
static __device__ __forceinline__ float bf2f(u16 h) {
  return __uint_as_float(((unsigned int)h) << 16);
}
static __device__ __forceinline__ unsigned int pk2(float a, float b) {
  return (unsigned int)f2bf(a) | ((unsigned int)f2bf(b) << 16);
}
static __device__ __forceinline__ v16bf mkfrag(v4u lo, v4u hi) {
  v8u t; t.lo = lo; t.hi = hi;
  return __builtin_bit_cast(v16bf, t);
}

// ---------------------------------------------------------------------------
// 16xN GEMM: A (16 x KD, bf16 rows in LDS) @ Bt (N x KD, bf16 rows in global,
// i.e. B pre-transposed so K is contiguous) -> 16 x (NT*16).
// Operand layout per CDNA5 ISA 7.12.2 (16-bit A 16x32; B stored as B^T with
// the same per-lane pattern). C layout: row = g + (lane>=16)*8, col = lane&15.
// Tile loop has a COMPILE-TIME trip count (NT/8 per wave, identical for all
// waves) so no exec-mask predication is generated around the WMMA body
// (ISA 7.12: EXEC must be all-ones for WMMA).
// EPI: 0 = +bias -> f32 LDS, 1 = +bias -> bf16 LDS, 2 = +bias, GELU -> bf16.
// ---------------------------------------------------------------------------
template <int KD, int NT, int EPI>
static __device__ __forceinline__ void gemm16(
    const u16* __restrict__ A, int lda,
    const u16* __restrict__ Bt,
    const float* __restrict__ bias,
    u16* out_bf, int ldo_bf,
    float* out_f, int ldo_f,
    int tid)
{
  static_assert(NT % 8 == 0, "tiles must divide evenly across 8 waves");
  const int lane = tid & 31;
  const int wave = tid >> 5;
  const int half = lane >> 4;
  const int l15  = lane & 15;
  const u16* abase = A + l15 * lda + half * 8;
#pragma unroll
  for (int t = 0; t < NT / 8; ++t) {
    const int nt = wave + t * 8;
    const u16* bbase = Bt + (size_t)(nt * 16 + l15) * KD + half * 8;
    v8f c = {};
#pragma unroll 4
    for (int k = 0; k < KD; k += 32) {
      v16bf a = mkfrag(*(const v4u*)(abase + k), *(const v4u*)(abase + k + 16));
      v16bf b = mkfrag(*(const v4u*)(bbase + k), *(const v4u*)(bbase + k + 16));
      c = __builtin_amdgcn_wmma_f32_16x16x32_bf16(false, a, false, b,
                                                  (short)0, c, false, false);
    }
    const int col = nt * 16 + l15;
    const float bs = bias[col];
#pragma unroll
    for (int g = 0; g < 8; ++g) {
      const int row = g + half * 8;
      float v = c[g] + bs;
      if constexpr (EPI == 2)
        v = 0.5f * v * (1.0f + erff(v * 0.70710678118654752f)); // exact GELU
      if constexpr (EPI == 0) out_f[row * ldo_f + col] = v;
      else                    out_bf[row * ldo_bf + col] = f2bf(v);
    }
  }
}

// ---------------------------------------------------------------------------
// Prep kernels (run every launch; deterministic, tiny cost)
// ---------------------------------------------------------------------------
__global__ void k_transpose_bf16(const float* __restrict__ W,
                                 u16* __restrict__ Wt, int K, int N) {
  int idx = blockIdx.x * 256 + threadIdx.x;   // idx = n*K + k
  if (idx >= K * N) return;
  int n = idx / K, k = idx - n * K;
  Wt[idx] = f2bf(W[k * N + n]);
}

__global__ void k_compute_Pt(const float* __restrict__ Wq,
                             const float* __restrict__ Wk,
                             u16* __restrict__ Pt) {
  int idx = blockIdx.x * 256 + threadIdx.x;   // idx = n*384 + k ; Pt[n][k]=P[k][n]
  if (idx >= D_IN * D_IN) return;
  int n = idx / D_IN, k = idx - n * D_IN;
  float s = 0.f;
  for (int a = 0; a < D_OUT; ++a) s += Wq[k * D_OUT + a] * Wk[n * D_OUT + a];
  Pt[idx] = f2bf(s);
}

__global__ void k_consts(const float* __restrict__ Wq, const float* __restrict__ bq,
                         const float* __restrict__ Wk, const float* __restrict__ bk,
                         float* __restrict__ u, float* __restrict__ u2,
                         float* __restrict__ t3) {
  int t = blockIdx.x * 256 + threadIdx.x;
  if (t < 384) {
    float s = 0.f;
    for (int a = 0; a < D_OUT; ++a) s += Wk[t * D_OUT + a] * bq[a];
    u[t] = s;
  } else if (t < 768) {
    int d = t - 384; float s = 0.f;
    for (int a = 0; a < D_OUT; ++a) s += Wq[d * D_OUT + a] * bk[a];
    u2[d] = s;
  } else if (t == 768) {
    float s = 0.f;
    for (int a = 0; a < D_OUT; ++a) s += bq[a] * bk[a];
    *t3 = s;
  }
}

// ---------------------------------------------------------------------------
// Main fused kernel: 2048 WGs x 256 threads, 16 windows (64 tokens) per WG.
// ---------------------------------------------------------------------------
__global__ __launch_bounds__(256)
void k_main(const float* __restrict__ x,
            const u16* __restrict__ Pt,  const u16* __restrict__ Wvt,
            const u16* __restrict__ W1t, const u16* __restrict__ W2t,
            const float* __restrict__ u,  const float* __restrict__ u2,
            const float* __restrict__ t3p,
            const float* __restrict__ bv, const float* __restrict__ b1,
            const float* __restrict__ b2,
            const float* __restrict__ gamma, const float* __restrict__ beta,
            float* __restrict__ yout)
{
  __shared__ __align__(16) char smem[LDS_BYTES];
  u16*   XTOK = (u16*)(smem + OFF_XTOK);
  u16*   XBAR = (u16*)(smem + OFF_XBAR);
  float* POUT = (float*)(smem + OFF_POUT);
  u16*   OUTR = (u16*)(smem + OFF_OUTR);
  float* CW   = (float*)(smem + OFF_CW);
  u16*   HDN  = (u16*)(smem + OFF_HDN);
  float* YROW = (float*)(smem + OFF_YROW);

  const int tid = threadIdx.x;
  const int wg  = blockIdx.x;

  // ---- Phase 1: gather 64 token rows (window layout), f32 -> bf16 -> LDS
  {
    int r    = tid >> 2;        // 0..63 token row
    int part = tid & 3;         // 96-col slice
    int lw = r >> 2, i = r & 3;
    int w  = wg * WPB + lw;
    int bt = w >> 8, n = w & 255;
    int gh = ((n >> 4) << 1) + (i >> 1);
    int gw = ((n & 15) << 1) + (i & 1);
    size_t grow = (size_t)bt * 1024 + (size_t)gh * 32 + gw;
    const v4f* src = (const v4f*)(x + grow * D_IN + part * 96);
    v4u* dst = (v4u*)(XTOK + r * KS384 + part * 96);
#pragma unroll
    for (int it = 0; it < 12; ++it) {
      v4f f0 = src[2 * it], f1 = src[2 * it + 1];
      v4u p;
      p.x = pk2(f0.x, f0.y); p.y = pk2(f0.z, f0.w);
      p.z = pk2(f1.x, f1.y); p.w = pk2(f1.z, f1.w);
      dst[it] = p;
    }
  }
  __syncthreads();

  // ---- Phase 2: window means (bf16 rows) + cw = m.u2 + t3
  {
    int row = tid >> 4, g16 = tid & 15;          // 16 threads per window
    const u16* x0 = XTOK + (4 * row + 0) * KS384;
    const u16* x1 = XTOK + (4 * row + 1) * KS384;
    const u16* x2 = XTOK + (4 * row + 2) * KS384;
    const u16* x3 = XTOK + (4 * row + 3) * KS384;
    u16* mrow = XBAR + row * KS384;
    float dot = 0.f;
#pragma unroll 4
    for (int j = 0; j < 24; ++j) {
      int c = g16 * 24 + j;
      float m = 0.25f * (bf2f(x0[c]) + bf2f(x1[c]) + bf2f(x2[c]) + bf2f(x3[c]));
      mrow[c] = f2bf(m);
      dot += m * u2[c];
    }
    for (int msk = 8; msk >= 1; msk >>= 1) dot += __shfl_xor(dot, msk, 16);
    if (g16 == 0) CW[row] = dot + *t3p;
  }
  __syncthreads();

  // ---- Phase 3: p = means @ P + u  (WMMA, f32 out)
  gemm16<384, 24, 0>(XBAR, KS384, Pt, u, nullptr, 0, POUT, FS384, tid);
  __syncthreads();

  // ---- Phase 4: logits = p.x_i + cw, softmax(4), xbar = sum a_i x_i
  {
    int lw = tid >> 4, j = tid & 15;
    const float* prow = POUT + lw * FS384;
    const u16* x0 = XTOK + (4 * lw + 0) * KS384;
    const u16* x1 = XTOK + (4 * lw + 1) * KS384;
    const u16* x2 = XTOK + (4 * lw + 2) * KS384;
    const u16* x3 = XTOK + (4 * lw + 3) * KS384;
    float d0 = 0.f, d1 = 0.f, d2 = 0.f, d3 = 0.f;
#pragma unroll 4
    for (int jj = 0; jj < 24; ++jj) {
      int c = j * 24 + jj;
      float pv = prow[c];
      d0 += pv * bf2f(x0[c]); d1 += pv * bf2f(x1[c]);
      d2 += pv * bf2f(x2[c]); d3 += pv * bf2f(x3[c]);
    }
    for (int msk = 8; msk >= 1; msk >>= 1) {
      d0 += __shfl_xor(d0, msk, 16); d1 += __shfl_xor(d1, msk, 16);
      d2 += __shfl_xor(d2, msk, 16); d3 += __shfl_xor(d3, msk, 16);
    }
    float cw = CW[lw];
    float l0 = (d0 + cw) * SCALE_QK, l1 = (d1 + cw) * SCALE_QK;
    float l2 = (d2 + cw) * SCALE_QK, l3 = (d3 + cw) * SCALE_QK;
    float mx = fmaxf(fmaxf(l0, l1), fmaxf(l2, l3));
    float e0 = __expf(l0 - mx), e1 = __expf(l1 - mx);
    float e2 = __expf(l2 - mx), e3 = __expf(l3 - mx);
    float inv = 1.0f / (e0 + e1 + e2 + e3);
    e0 *= inv; e1 *= inv; e2 *= inv; e3 *= inv;
    u16* xb = XBAR + lw * KS384;                  // overwrite means
#pragma unroll 4
    for (int jj = 0; jj < 24; ++jj) {
      int c = j * 24 + jj;
      float v = e0 * bf2f(x0[c]) + e1 * bf2f(x1[c]) +
                e2 * bf2f(x2[c]) + e3 * bf2f(x3[c]);
      xb[c] = f2bf(v);
    }
  }
  __syncthreads();

  // ---- Phase 5: out = xbar @ Wv + bv  (bf16 out)
  gemm16<384, 24, 1>(XBAR, KS384, Wvt, bv, OUTR, KS384, nullptr, 0, tid);
  __syncthreads();

  // ---- Phase 6: hdn = GELU(out @ W1 + b1)  (bf16 out, overlays XTOK)
  gemm16<384, 48, 2>(OUTR, KS384, W1t, b1, HDN, KS768, nullptr, 0, tid);
  __syncthreads();

  // ---- Phase 7: y = hdn @ W2 + b2  (f32 out)
  gemm16<768, 24, 0>(HDN, KS768, W2t, b2, nullptr, 0, YROW, FS384, tid);
  __syncthreads();

  // ---- Phase 8: LayerNorm + store
  {
    const int lane = tid & 31, wave = tid >> 5;
    for (int r = wave; r < WPB; r += 8) {
      const float* yr = YROW + r * FS384;
      float s = 0.f, s2 = 0.f;
#pragma unroll
      for (int i = 0; i < 12; ++i) {
        float v = yr[lane + 32 * i];
        s += v; s2 += v * v;
      }
      for (int msk = 16; msk >= 1; msk >>= 1) {
        s += __shfl_xor(s, msk, 32); s2 += __shfl_xor(s2, msk, 32);
      }
      float mean = s * (1.0f / 384.0f);
      float var  = s2 * (1.0f / 384.0f) - mean * mean;
      float rstd = rsqrtf(var + 1e-5f);
      float* orow = yout + (size_t)(wg * WPB + r) * D_OUT;
#pragma unroll
      for (int i = 0; i < 12; ++i) {
        int c = lane + 32 * i;
        orow[c] = (yr[c] - mean) * rstd * gamma[c] + beta[c];
      }
    }
  }
}

// ---------------------------------------------------------------------------
extern "C" void kernel_launch(void* const* d_in, const int* in_sizes, int n_in,
                              void* d_out, int out_size, void* d_ws, size_t ws_size,
                              hipStream_t stream) {
  (void)in_sizes; (void)n_in; (void)out_size; (void)ws_size;
  const float* x     = (const float*)d_in[0];
  const float* Wq    = (const float*)d_in[1];
  const float* bq    = (const float*)d_in[2];
  const float* Wk    = (const float*)d_in[3];
  const float* bk    = (const float*)d_in[4];
  const float* Wv    = (const float*)d_in[5];
  const float* bv    = (const float*)d_in[6];
  const float* W1    = (const float*)d_in[7];
  const float* b1    = (const float*)d_in[8];
  const float* W2    = (const float*)d_in[9];
  const float* b2    = (const float*)d_in[10];
  const float* gamma = (const float*)d_in[11];
  const float* beta  = (const float*)d_in[12];
  float* out = (float*)d_out;

  char* ws = (char*)d_ws;
  u16*   Pt  = (u16*)(ws + 0);          // 384x384 bf16
  u16*   Wvt = (u16*)(ws + 294912);     // 384x384 bf16
  u16*   W1t = (u16*)(ws + 589824);     // 768x384 bf16
  u16*   W2t = (u16*)(ws + 1179648);    // 384x768 bf16
  float* u   = (float*)(ws + 1769472);  // 384 f32
  float* u2  = (float*)(ws + 1771008);  // 384 f32
  float* t3  = (float*)(ws + 1772544);  // 1 f32

  k_transpose_bf16<<<(384 * 384 + 255) / 256, 256, 0, stream>>>(Wv, Wvt, 384, 384);
  k_transpose_bf16<<<(384 * 768 + 255) / 256, 256, 0, stream>>>(W1, W1t, 384, 768);
  k_transpose_bf16<<<(768 * 384 + 255) / 256, 256, 0, stream>>>(W2, W2t, 768, 384);
  k_compute_Pt<<<(384 * 384 + 255) / 256, 256, 0, stream>>>(Wq, Wk, Pt);
  k_consts<<<4, 256, 0, stream>>>(Wq, bq, Wk, bk, u, u2, t3);

  k_main<<<NWG, 256, 0, stream>>>(x, Pt, Wvt, W1t, W2t, u, u2, t3,
                                  bv, b1, b2, gamma, beta, out);
}
